// RoseNet_body_549755814488
// MI455X (gfx1250) — matrix operand order repaired and checked by reading, not compiled, not run on hardware
//
#include <hip/hip_runtime.h>
#include <hip/hip_bf16.h>
#include <math.h>

typedef __attribute__((ext_vector_type(16))) _Float16 v16h;
typedef __attribute__((ext_vector_type(8)))  _Float16 v8h;
typedef __attribute__((ext_vector_type(8)))  float    v8f;

#define BB   512
#define C1H  64
#define CC   128
#define HH   15
#define WW   15
#define HW   225
#define CRR  32
#define CIDX 112
#define EPS_BN  1e-5f
#define EPS_COS 1e-8f

// Async global->LDS staging path (CDNA5 GLOBAL_LOAD_ASYNC_TO_LDS, ASYNCcnt).
// The builtin is declared by this toolchain; its first parameter is a global
// (addrspace(1)) int*, second an LDS (addrspace(3)) int* (per hipcc diagnostic).
#if defined(__has_builtin)
#  if __has_builtin(__builtin_amdgcn_global_load_async_to_lds_b32)
#    define HAS_ASYNC_LDS 1
#  endif
#endif
#ifndef HAS_ASYNC_LDS
#  define HAS_ASYNC_LDS 0
#endif

typedef __attribute__((address_space(1))) int* gint_p;
typedef __attribute__((address_space(3))) int* lint_p;

__device__ __forceinline__ float sigmoidf_(float x){ return 1.0f/(1.0f+__expf(-x)); }

// ---------------- block reductions (blockDim must be power of two) -------------
__device__ __forceinline__ float blk_red_sum(float v, float* red){
  int t = threadIdx.x;
  red[t] = v; __syncthreads();
  for (int s = blockDim.x>>1; s > 0; s >>= 1){
    if (t < s) red[t] += red[t+s];
    __syncthreads();
  }
  float r = red[0]; __syncthreads();
  return r;
}
__device__ __forceinline__ float blk_red_max(float v, float* red){
  int t = threadIdx.x;
  red[t] = v; __syncthreads();
  for (int s = blockDim.x>>1; s > 0; s >>= 1){
    if (t < s) red[t] = fmaxf(red[t], red[t+s]);
    __syncthreads();
  }
  float r = red[0]; __syncthreads();
  return r;
}

// ---------------- attention #1: concat + dist/cos softmax + residual ----------
__global__ void attn1_kernel(const float* __restrict__ x1, const float* __restrict__ x2,
                             const float* __restrict__ lam, float* __restrict__ out)
{
  __shared__ float cent[CC];
  __shared__ float qn[HW];
  __shared__ float sim_e[HW];
  __shared__ float sim_c[HW];
  __shared__ float att[HW];
  __shared__ float red[256];
  int b = blockIdx.x, t = threadIdx.x;
  const float* xb1 = x1 + (size_t)b*C1H*HW;
  const float* xb2 = x2 + (size_t)b*C1H*HW;
  if (t < CC) cent[t] = (t < C1H) ? xb1[t*HW + CIDX] : xb2[(t-C1H)*HW + CIDX];
  __syncthreads();
  float dt = 0.0f;
  if (t < HW){
    float se = 0.0f, sq = 0.0f;
    for (int c = 0; c < CC; ++c){
      float v = (c < C1H) ? xb1[c*HW + t] : xb2[(c-C1H)*HW + t];
      float d = v - cent[c];
      se += d*d; sq += v*v; dt += cent[c]*v;
    }
    qn[t] = sqrtf(sq);
    sim_e[t] = 1.0f/(1.0f + sqrtf(se));
  }
  __syncthreads();
  float cn = qn[CIDX];
  if (t < HW)
    sim_c[t] = dt / (fmaxf(cn, EPS_COS) * fmaxf(qn[t], EPS_COS));
  __syncthreads();
  float ve = (t < HW) ? sim_e[t] : -1e30f;
  float me = blk_red_max(ve, red);
  float ee = (t < HW) ? expf(ve - me) : 0.0f;
  float sse = blk_red_sum(ee, red);
  float vc = (t < HW) ? sim_c[t] : -1e30f;
  float mc = blk_red_max(vc, red);
  float ec = (t < HW) ? expf(vc - mc) : 0.0f;
  float ssc = blk_red_sum(ec, red);
  float lmd = sigmoidf_(lam[0]);
  if (t < HW) att[t] = lmd*(ee/sse) + (1.0f-lmd)*(ec/ssc);
  __syncthreads();
  float* ob = out + (size_t)b*CC*HW;
  for (int i = t; i < CC*HW; i += blockDim.x){
    int c = i / HW, p = i % HW;
    float v = (c < C1H) ? xb1[c*HW + p] : xb2[(c-C1H)*HW + p];
    ob[i] = (1.0f + att[p]) * v;
  }
}

// ---------------- attention #2: qkv distance softmax + residual ---------------
__global__ void attn2_kernel(const float* __restrict__ qkv, const float* __restrict__ xres,
                             float* __restrict__ out)
{
  __shared__ float cent[CC];
  __shared__ float simv[HW];
  __shared__ float att[HW];
  __shared__ float red[256];
  int b = blockIdx.x, t = threadIdx.x;
  const float* qb = qkv + (size_t)b*3*CC*HW;
  if (t < CC) cent[t] = qb[t*HW + CIDX];
  __syncthreads();
  if (t < HW){
    float se = 0.0f;
    const float* kb = qb + CC*HW;
    for (int c = 0; c < CC; ++c){ float d = cent[c] - kb[c*HW + t]; se += d*d; }
    simv[t] = 1.0f/(1.0f + sqrtf(se));
  }
  __syncthreads();
  float v = (t < HW) ? simv[t] : -1e30f;
  float m = blk_red_max(v, red);
  float e = (t < HW) ? expf(v - m) : 0.0f;
  float s = blk_red_sum(e, red);
  if (t < HW) att[t] = e / s;
  __syncthreads();
  const float* vb = qb + 2*CC*HW;
  const float* rb = xres + (size_t)b*CC*HW;
  float* ob = out + (size_t)b*CC*HW;
  for (int i = t; i < CC*HW; i += blockDim.x){
    int p = i % HW;
    ob[i] = att[p]*vb[i] + rb[i];
  }
}

// ---------------- batchnorm stats -> scale/shift per channel ------------------
__global__ void bn_stats_kernel(const float* __restrict__ x, int nchan,
                                const float* __restrict__ gamma, const float* __restrict__ beta,
                                float* __restrict__ scale, float* __restrict__ shift)
{
  __shared__ float red[256];
  int c = blockIdx.x, t = threadIdx.x;
  float s = 0.0f, sq = 0.0f;
  const int N = BB*HW;
  for (int i = t; i < N; i += blockDim.x){
    int b = i / HW, p = i % HW;
    float v = x[((size_t)b*nchan + c)*HW + p];
    s += v; sq += v*v;
  }
  s  = blk_red_sum(s, red);
  sq = blk_red_sum(sq, red);
  if (t == 0){
    float m    = s  / (float)N;
    float var  = sq / (float)N - m*m;
    float rstd = rsqrtf(var + EPS_BN);
    float sc   = rstd * gamma[c];
    scale[c] = sc;
    shift[c] = beta[c] - m*sc;
  }
}

// ---------------- fold BN into a 1x1 conv (Cin == 128) ------------------------
__global__ void fold_pw_kernel(const float* __restrict__ W, const float* __restrict__ scale,
                               const float* __restrict__ shift, _Float16* __restrict__ Wf,
                               float* __restrict__ biasF)
{
  __shared__ float red[CC];
  int o = blockIdx.x, c = threadIdx.x;     // blockDim = 128
  float w = W[o*CC + c];
  Wf[o*CC + c] = (_Float16)(w * scale[c]);
  red[c] = w * shift[c];
  __syncthreads();
  for (int s = CC>>1; s > 0; s >>= 1){
    if (c < s) red[c] += red[c+s];
    __syncthreads();
  }
  if (c == 0) biasF[o] = red[0];
}

// ---------------- plain f32 -> f16 weight conversion --------------------------
__global__ void cvt_f16_kernel(const float* __restrict__ src, _Float16* __restrict__ dst, int n){
  int i = blockIdx.x*blockDim.x + threadIdx.x;
  if (i < n) dst[i] = (_Float16)src[i];
}

// ---------------- (Cout,Cin,7,7) f32 -> [49][Cout][Cin] f16 -------------------
__global__ void prep_convw_kernel(const float* __restrict__ src, _Float16* __restrict__ dst,
                                  int Cout, int Cin){
  int i = blockIdx.x*blockDim.x + threadIdx.x;
  int n = Cout*Cin*49;
  if (i < n){
    int o = i / (Cin*49);
    int r = i % (Cin*49);
    int c = r / 49, pos = r % 49;
    dst[((size_t)pos*Cout + o)*Cin + c] = (_Float16)src[i];
  }
}

// ---------------- WMMA GEMM: Out[b,m,p] = epi(sum_k A[m,k] * X[b,k,p]) --------
// A: f16 row-major [M][K]; X: f32 [B][K][HW].
// Block = PT pixel tiles x (M/16) M tiles; one wave per (M tile, pixel tile).
// K is a template constant (32 or 128) so the K loop fully unrolls and all
// fragment loads are hoisted ahead of the WMMA chain.
template<int EPI, int K, int PT>
__global__ void wmma_gemm_kernel(const _Float16* __restrict__ A, const float* __restrict__ X,
                                 float* __restrict__ Out,
                                 const float* __restrict__ bias,
                                 const float* __restrict__ pa, const float* __restrict__ pb,
                                 int M)
{
  constexpr int NP = PT*16;                       // pixels staged per block
  __shared__ _Float16 sB[NP*K];                   // [local pixel][K], K contiguous
  int b   = blockIdx.y;
  int pB  = blockIdx.x * NP;                      // first pixel of block
  int tid = threadIdx.x;
  const float* Xb = X + (size_t)b*K*HW;
  for (int i = tid; i < NP*K; i += blockDim.x){
    int k = i / NP, pp = i % NP;                  // consecutive tid -> consecutive pixel
    int p = pB + pp;
    float v = (p < HW) ? Xb[k*HW + p] : 0.0f;
    sB[pp*K + k] = (_Float16)v;
  }
  __syncthreads();

  int wave = tid >> 5, lane = tid & 31;
  int mT    = M >> 4;
  int mTile = wave % mT;
  int pt    = wave / mT;
  int p0    = pB + pt*16;
  if (p0 >= HW) return;                           // wave-uniform; no later syncs
  int lr = lane & 15, hi = lane >> 4;
  v8f acc; for (int r = 0; r < 8; ++r) acc[r] = 0.0f;
  const _Float16* Arow = A + (size_t)(mTile*16 + lr)*K;
  const _Float16* Brow = &sB[(pt*16 + lr)*K];

#pragma unroll
  for (int kk = 0; kk < K; kk += 32){
    v8h a0 = *(const v8h*)(Arow + kk + hi*16);
    v8h a1 = *(const v8h*)(Arow + kk + hi*16 + 8);
    v16h af; for (int i = 0; i < 8; ++i){ af[i] = a0[i]; af[i+8] = a1[i]; }
    v16h bf = *(const v16h*)(Brow + kk + hi*16);
    acc = __builtin_amdgcn_wmma_f32_16x16x32_f16(false, af, false, bf,
                                                 (short)0, acc, false, false);
  }

  int p = p0 + lr;
  if (p < HW){
    float* ob = Out + (size_t)b*M*HW;
    for (int r = 0; r < 8; ++r){
      int m = mTile*16 + r + hi*8;
      float v = acc[r];
      if      (EPI == 1){ v += bias[m]; v = (v >= 0.f) ? v : 0.01f*v;
                          v = v*pa[m] + pb[m]; v = fmaxf(v, 0.f); }
      else if (EPI == 2){ v += bias[m]; v = (v >= 0.f) ? v : 0.01f*v; }
      else if (EPI == 3){ v += bias[m]; v = fmaxf(v, 0.f); }
      else if (EPI == 4){ v += bias[m]; v = sigmoidf_(v); }
      ob[m*HW + p] = v;
    }
  }
}

// ---------------- WMMA implicit-GEMM 7x7 conv (pad 3), zero-halo LDS ----------
// wA: f16 [49][Cout][Cin]; X: f32 [B][Cin][HW]; Out: f32 [B][Cout][HW].
// The whole image is staged once per block into a 21x21 zero-padded halo image,
// making every shifted tap an unconditional contiguous 32B LDS read (branchless
// inner loop, EXEC stays all-ones). Block covers PT pixel tiles x all M tiles.
template<int CIN, int PT>
__global__ void wmma_conv7_kernel(const _Float16* __restrict__ wA, const float* __restrict__ X,
                                  const float* __restrict__ bias, float* __restrict__ Out,
                                  int Cout)
{
  constexpr int PW = HH + 6;                      // 21 padded rows/cols
  extern __shared__ _Float16 sX[];                // [PW*PW][CIN], CIN contiguous
  int b   = blockIdx.y;
  int tid = threadIdx.x;
  const float* Xb = X + (size_t)b*CIN*HW;
  for (int i = tid; i < PW*PW*CIN; i += blockDim.x){
    int c  = i / (PW*PW);
    int rc = i % (PW*PW);
    int r  = rc / PW, cl = rc % PW;
    int yy = r - 3, xx = cl - 3;
    float v = (yy >= 0 && yy < HH && xx >= 0 && xx < WW) ? Xb[c*HW + yy*WW + xx] : 0.0f;
    sX[rc*CIN + c] = (_Float16)v;
  }
  __syncthreads();

  int wave = tid >> 5, lane = tid & 31;
  int oT    = Cout >> 4;
  int oTile = wave % oT;
  int pt    = wave / oT;
  int p0    = (blockIdx.x*PT + pt)*16;
  if (p0 >= HW) return;                           // wave-uniform; no later syncs
  int lr = lane & 15, hi = lane >> 4;
  v8f acc; for (int r = 0; r < 8; ++r) acc[r] = 0.0f;
  int pn = p0 + lr; if (pn > HW-1) pn = HW-1;     // clamp; stores are guarded
  int y0 = pn / WW, x0 = pn % WW;
  int cellBase = y0*PW + x0;                      // pos offsets: (pos/7)*PW + pos%7

  for (int pos = 0; pos < 49; ++pos){
    const _Float16* wrow = wA + ((size_t)pos*Cout + oTile*16 + lr)*CIN;
    if (pos + 1 < 49)
      __builtin_prefetch(wA + ((size_t)(pos+1)*Cout + oTile*16 + lr)*CIN, 0, 0);
    int ps = (cellBase + (pos/7)*PW + (pos%7))*CIN;
#pragma unroll
    for (int cb = 0; cb < CIN; cb += 32){
      v16h bf = *(const v16h*)(&sX[ps + cb + hi*16]);
      v8h a0 = *(const v8h*)(wrow + cb + hi*16);
      v8h a1 = *(const v8h*)(wrow + cb + hi*16 + 8);
      v16h af; for (int i = 0; i < 8; ++i){ af[i] = a0[i]; af[i+8] = a1[i]; }
      acc = __builtin_amdgcn_wmma_f32_16x16x32_f16(false, af, false, bf,
                                                   (short)0, acc, false, false);
    }
  }

  int p = p0 + lr;
  if (p < HW){
    float* ob = Out + (size_t)b*Cout*HW;
    for (int r = 0; r < 8; ++r){
      int m = oTile*16 + r + hi*8;
      ob[m*HW + p] = acc[r] + bias[m];
    }
  }
}

// ---------------- depthwise 3x3 + bias + relu ---------------------------------
// Plane staging uses CDNA5 async global->LDS copy (tracked by ASYNCcnt, drained
// with s_wait_asynccnt before the barrier).
__global__ void dw3x3_kernel(const float* __restrict__ X, const float* __restrict__ dW,
                             const float* __restrict__ db, float* __restrict__ Out)
{
  __shared__ float plane[HW];
  int bc = blockIdx.x;
  int c = bc % CC;
  int t = threadIdx.x;
  const float* xb = X + (size_t)bc*HW;
#if HAS_ASYNC_LDS
  if (t < HW){
    __builtin_amdgcn_global_load_async_to_lds_b32(
        (gint_p)(xb + t),
        (lint_p)(&plane[t]),
        0, 0);
  }
  asm volatile("s_wait_asynccnt 0x0" ::: "memory");
  __syncthreads();
#else
  if (t < HW) plane[t] = xb[t];
  __syncthreads();
#endif
  if (t < HW){
    int y = t / WW, x = t % WW;
    float acc = db[c];
    for (int ky = 0; ky < 3; ++ky)
      for (int kx = 0; kx < 3; ++kx){
        int yy = y + ky - 1, xx = x + kx - 1;
        if (yy >= 0 && yy < HH && xx >= 0 && xx < WW)
          acc += dW[c*9 + ky*3 + kx] * plane[yy*WW + xx];
      }
    Out[(size_t)bc*HW + t] = fmaxf(acc, 0.f);
  }
}

// ---------------- per-channel affine (+relu) elementwise ----------------------
template<int RELU>
__global__ void bn_apply_kernel(const float* __restrict__ X, float* __restrict__ Out,
                                const float* __restrict__ scale, const float* __restrict__ shift,
                                int nchan, int n)
{
  int i = blockIdx.x*blockDim.x + threadIdx.x;
  if (i < n){
    int c = (i / HW) % nchan;
    float v = X[i]*scale[c] + shift[c];
    if (RELU) v = fmaxf(v, 0.f);
    Out[i] = v;
  }
}

__global__ void mul_kernel(const float* __restrict__ a, const float* __restrict__ g,
                           float* __restrict__ o, int n){
  int i = blockIdx.x*blockDim.x + threadIdx.x;
  if (i < n) o[i] = a[i]*g[i];
}

__global__ void final_kernel(const float* __restrict__ xca, const float* __restrict__ s2,
                             const float* __restrict__ scale, const float* __restrict__ shift,
                             float* __restrict__ out, int n)
{
  int i = blockIdx.x*blockDim.x + threadIdx.x;
  if (i < n){
    int c = (i / HW) % CC;
    out[i] = xca[i] * sigmoidf_(s2[i]*scale[c] + shift[c]);
  }
}

// =============================================================================
extern "C" void kernel_launch(void* const* d_in, const int* in_sizes, int n_in,
                              void* d_out, int out_size, void* d_ws, size_t ws_size,
                              hipStream_t stream)
{
  (void)in_sizes; (void)n_in; (void)out_size; (void)ws_size;

  const float* x1    = (const float*)d_in[0];
  const float* x2i   = (const float*)d_in[1];
  const float* lam   = (const float*)d_in[2];
  const float* bn1_g = (const float*)d_in[3];
  const float* bn1_b = (const float*)d_in[4];
  const float* pW1   = (const float*)d_in[5];
  const float* dW1   = (const float*)d_in[6];
  const float* db1   = (const float*)d_in[7];
  const float* bn2_g = (const float*)d_in[8];
  const float* bn2_b = (const float*)d_in[9];
  const float* pW2   = (const float*)d_in[10];
  const float* dW2   = (const float*)d_in[11];
  const float* db2   = (const float*)d_in[12];
  const float* qkv_W = (const float*)d_in[13];
  const float* bn3_g = (const float*)d_in[14];
  const float* bn3_b = (const float*)d_in[15];
  const float* ca_w1 = (const float*)d_in[16];
  const float* ca_b1 = (const float*)d_in[17];
  const float* ca_w2 = (const float*)d_in[18];
  const float* ca_b2 = (const float*)d_in[19];
  const float* sa_w1 = (const float*)d_in[20];
  const float* sa_b1 = (const float*)d_in[21];
  const float* sbn1_g= (const float*)d_in[22];
  const float* sbn1_b= (const float*)d_in[23];
  const float* sa_w2 = (const float*)d_in[24];
  const float* sa_b2 = (const float*)d_in[25];
  const float* sbn2_g= (const float*)d_in[26];
  const float* sbn2_b= (const float*)d_in[27];
  float* out = (float*)d_out;

  const size_t SZT = (size_t)BB*CC*HW;           // 14,745,600 elems
  char* w = (char*)d_ws;
  float* bufA = (float*)w;
  float* bufB = (float*)(w + SZT*4);
  float* qbuf = (float*)(w + 2*SZT*4);           // B*3C*HW f32 (also reused for h/s1)
  char*  sm   = w + 2*SZT*4 + (size_t)BB*3*CC*HW*4;

  _Float16* W1f   = (_Float16*)sm; sm += (size_t)CC*CC*2;
  _Float16* W2f   = (_Float16*)sm; sm += (size_t)CC*CC*2;
  _Float16* Wqkvf = (_Float16*)sm; sm += (size_t)3*CC*CC*2;
  _Float16* Wca1f = (_Float16*)sm; sm += (size_t)CRR*CC*2;
  _Float16* Wca2f = (_Float16*)sm; sm += (size_t)CC*CRR*2;
  _Float16* wA1   = (_Float16*)sm; sm += (size_t)49*CRR*CC*2;
  _Float16* wA2   = (_Float16*)sm; sm += (size_t)49*CC*CRR*2;
  float* bias1 = (float*)sm; sm += CC*4;
  float* bias2 = (float*)sm; sm += CC*4;
  float* sc1   = (float*)sm; sm += CC*4;
  float* sh1   = (float*)sm; sm += CC*4;
  float* sc2   = (float*)sm; sm += CC*4;
  float* sh2   = (float*)sm; sm += CC*4;
  float* sc3   = (float*)sm; sm += CC*4;
  float* sh3   = (float*)sm; sm += CC*4;
  float* scs1  = (float*)sm; sm += CRR*4;
  float* shs1  = (float*)sm; sm += CRR*4;
  float* scs2  = (float*)sm; sm += CC*4;
  float* shs2  = (float*)sm; sm += CC*4;
  float* hbuf  = qbuf;                           // B*CRR*HW reuse inside qkv region

  const int EW = 256;
  const int nEl = (int)SZT;
  const int PWH = HH + 6;                        // 21

  // 1) concat + attention1 + residual -> bufA
  attn1_kernel<<<dim3(BB), dim3(256), 0, stream>>>(x1, x2i, lam, bufA);

  // 2) BN1 stats + fold into pW1 -> W1f/bias1
  bn_stats_kernel<<<dim3(CC), dim3(256), 0, stream>>>(bufA, CC, bn1_g, bn1_b, sc1, sh1);
  fold_pw_kernel<<<dim3(CC), dim3(CC), 0, stream>>>(pW1, sc1, sh1, W1f, bias1);

  // 3) gemm1 fused (pw1 + leaky + dw1x1 + relu) -> bufB   (M=128,K=128,PT=1)
  wmma_gemm_kernel<1,128,1><<<dim3(15, BB), dim3(256), 0, stream>>>(
      W1f, bufA, bufB, bias1, dW1, db1, CC);

  // 4) BN2 stats + fold into pW2 -> W2f/bias2
  bn_stats_kernel<<<dim3(CC), dim3(256), 0, stream>>>(bufB, CC, bn2_g, bn2_b, sc2, sh2);
  fold_pw_kernel<<<dim3(CC), dim3(CC), 0, stream>>>(pW2, sc2, sh2, W2f, bias2);

  // 5) gemm2 (pw2 + leaky) -> bufA ; depthwise 3x3 + relu -> bufB
  wmma_gemm_kernel<2,128,1><<<dim3(15, BB), dim3(256), 0, stream>>>(
      W2f, bufB, bufA, bias2, nullptr, nullptr, CC);
  dw3x3_kernel<<<dim3(BB*CC), dim3(256), 0, stream>>>(bufA, dW2, db2, bufB);

  // 6) qkv projection (M=384) -> qbuf ; attention2 + residual -> bufA
  cvt_f16_kernel<<<dim3((3*CC*CC + 255)/256), dim3(256), 0, stream>>>(qkv_W, Wqkvf, 3*CC*CC);
  wmma_gemm_kernel<0,128,1><<<dim3(15, BB), dim3(768), 0, stream>>>(
      Wqkvf, bufB, qbuf, nullptr, nullptr, nullptr, 3*CC);
  attn2_kernel<<<dim3(BB), dim3(256), 0, stream>>>(qbuf, bufB, bufA);

  // 7) BN3 -> bufB (xbn)
  bn_stats_kernel<<<dim3(CC), dim3(256), 0, stream>>>(bufA, CC, bn3_g, bn3_b, sc3, sh3);
  bn_apply_kernel<0><<<dim3((nEl + EW - 1)/EW), dim3(EW), 0, stream>>>(
      bufA, bufB, sc3, sh3, CC, nEl);

  // 8) channel attention MLP: h = relu(W1@xbn+b1), g = sigmoid(W2@h+b2), xca = xbn*g
  cvt_f16_kernel<<<dim3((CRR*CC + 255)/256), dim3(256), 0, stream>>>(ca_w1, Wca1f, CRR*CC);
  cvt_f16_kernel<<<dim3((CC*CRR + 255)/256), dim3(256), 0, stream>>>(ca_w2, Wca2f, CC*CRR);
  wmma_gemm_kernel<3,128,4><<<dim3(4, BB), dim3(256), 0, stream>>>(   // M=32, 4 ptiles/block
      Wca1f, bufB, hbuf, ca_b1, nullptr, nullptr, CRR);
  wmma_gemm_kernel<4,32,1><<<dim3(15, BB), dim3(256), 0, stream>>>(   // M=128, K=32
      Wca2f, hbuf, bufA, ca_b2, nullptr, nullptr, CC);
  mul_kernel<<<dim3((nEl + EW - 1)/EW), dim3(EW), 0, stream>>>(bufB, bufA, bufA, nEl);

  // 9) spatial attention conv1 (7x7, 128->32) -> hbuf ; BN + relu in place
  prep_convw_kernel<<<dim3((CRR*CC*49 + 255)/256), dim3(256), 0, stream>>>(sa_w1, wA1, CRR, CC);
  wmma_conv7_kernel<CC,4><<<dim3(4, BB), dim3(256), (size_t)PWH*PWH*CC*2, stream>>>(
      wA1, bufA, sa_b1, hbuf, CRR);               // 2 oTiles x 4 ptiles = 8 waves
  bn_stats_kernel<<<dim3(CRR), dim3(256), 0, stream>>>(hbuf, CRR, sbn1_g, sbn1_b, scs1, shs1);
  {
    int n1 = BB*CRR*HW;
    bn_apply_kernel<1><<<dim3((n1 + EW - 1)/EW), dim3(EW), 0, stream>>>(
        hbuf, hbuf, scs1, shs1, CRR, n1);
  }

  // 10) spatial attention conv2 (7x7, 32->128) -> bufB ; BN stats ; final gating
  prep_convw_kernel<<<dim3((CC*CRR*49 + 255)/256), dim3(256), 0, stream>>>(sa_w2, wA2, CC, CRR);
  wmma_conv7_kernel<CRR,2><<<dim3(8, BB), dim3(512), (size_t)PWH*PWH*CRR*2, stream>>>(
      wA2, hbuf, sa_b2, bufB, CC);                // 8 oTiles x 2 ptiles = 16 waves
  bn_stats_kernel<<<dim3(CC), dim3(256), 0, stream>>>(bufB, CC, sbn2_g, sbn2_b, scs2, shs2);
  final_kernel<<<dim3((nEl + EW - 1)/EW), dim3(EW), 0, stream>>>(
      bufA, bufB, scs2, shs2, out, nEl);
}
